// MAATModel_62070867361934
// MI455X (gfx1250) — compile-verified, matrix operand order checked by reading, not run on hardware
//
#include <hip/hip_runtime.h>
#include <hip/hip_bf16.h>

typedef __attribute__((ext_vector_type(16))) _Float16 v16h;
typedef __attribute__((ext_vector_type(8)))  float    v8f;
typedef __attribute__((ext_vector_type(4)))  float    v4f;

// global (address_space(1)) pointer types: force global_load/global_store,
// so VMEM is tracked by LOADcnt/STOREcnt only (flat also ties up DScnt).
typedef const __attribute__((address_space(1))) float  GCF;
typedef       __attribute__((address_space(1))) float  GF;
typedef const __attribute__((address_space(1))) v4f    GCV4;

#define DMODEL 512
#define NB     8      // batch
#define LLEN   512    // sequence length
#define NH     8      // heads
#define EHD    64     // head dim
#define DINNER 768
#define DSTATE 8
#define NLAY   3
#define ENCIN  38
#define COUT   38
#define MROWS  4096   // NB*LLEN

__device__ __forceinline__ float act_apply(float v, int act) {
    if (act == 1) return 0.5f * v * (1.f + erff(v * 0.7071067811865475f));   // exact gelu
    if (act == 2) return v / (1.f + expf(-v));                               // silu
    if (act == 3) return 1.f / (1.f + expf(-v));                             // sigmoid
    return v;
}

// Contiguous-k fragment load (ISA 16-bit layout). p points at row base + k0 + kb,
// where kb = (lane>=16)*8. Elements e=0..7 -> k +0..7 ; e=8..15 -> k +16..23.
// Two 32-byte runs -> 4x global_load_b128 + packed f32->f16 converts.
__device__ __forceinline__ v16h load_frag_c(GCF* p)
{
    v4f a = ((GCV4*)p)[0];
    v4f b = ((GCV4*)p)[1];
    v4f c = ((GCV4*)(p + 16))[0];
    v4f d = ((GCV4*)(p + 16))[1];
    v16h f;
    f[0]  = (_Float16)a.x; f[1]  = (_Float16)a.y; f[2]  = (_Float16)a.z; f[3]  = (_Float16)a.w;
    f[4]  = (_Float16)b.x; f[5]  = (_Float16)b.y; f[6]  = (_Float16)b.z; f[7]  = (_Float16)b.w;
    f[8]  = (_Float16)c.x; f[9]  = (_Float16)c.y; f[10] = (_Float16)c.z; f[11] = (_Float16)c.w;
    f[12] = (_Float16)d.x; f[13] = (_Float16)d.y; f[14] = (_Float16)d.z; f[15] = (_Float16)d.w;
    return f;
}

// Strided-k fragment load (only the attn@V GEMM uses this: B = V with k-stride 512).
// p points at row base + (k0+kb)*stride; independent offsets e*stride.
__device__ __forceinline__ v16h load_frag_s(GCF* p, long long stride)
{
    v16h f;
#pragma unroll
    for (int e = 0; e < 8; ++e) {
        f[e]     = (_Float16)p[(long long)e * stride];
        f[e + 8] = (_Float16)p[(long long)(e + 16) * stride];
    }
    return f;
}

// ---------------------------------------------------------------------------
// Generic batched WMMA GEMM: C[m][n] = act(alpha * sum_k A[m,k]*B[n,k] + bias[n])
// One wave32 per 32x64 C tile: 2 A-frags x 4 B-frags -> 8 WMMAs per K-step.
// A k-stride is always 1; B k-stride is 1 when CONTIG_B (all weight GEMMs +
// scores), else sBk (attn@V). Batch z: off = (z/zdiv)*sXo + (z%zdiv)*sXi.
// Requires: M % 32 == 0, N % 64 == 0, K % 32 == 0 (true for all uses here).
// ---------------------------------------------------------------------------
template <bool CONTIG_B>
__launch_bounds__(32)
__global__ void wmma_gemm_kernel(const float* __restrict__ A, const float* __restrict__ Bm,
                                 const float* __restrict__ bias, float* __restrict__ C,
                                 int K, int sAm, int sBn, int sBk, int sCm,
                                 long long sAo, long long sAi,
                                 long long sBo, long long sBi,
                                 long long sCo, long long sCi,
                                 int zdiv, float alpha, int act)
{
    int lane = threadIdx.x & 31;
    int tn = blockIdx.x, tm = blockIdx.y, z = blockIdx.z;
    int zo = z / zdiv, zi = z % zdiv;
    const float* Ab = A  + (long long)zo * sAo + (long long)zi * sAi;
    const float* Bb = Bm + (long long)zo * sBo + (long long)zi * sBi;
    long long offC  = (long long)zo * sCo + (long long)zi * sCi;

    int kb = (lane >> 4) << 3;           // lanes 16-31 hold K+8 within each 16-group

    // scalar per-fragment global pointers (explicit AS1 -> global_load_b128)
    GCF* a0 = (GCF*)(Ab + (long long)(tm * 32 +  0 + (lane & 15)) * sAm + kb);
    GCF* a1 = (GCF*)(Ab + (long long)(tm * 32 + 16 + (lane & 15)) * sAm + kb);
    long long bOff0 = CONTIG_B ? (long long)kb : (long long)kb * sBk;
    GCF* b0 = (GCF*)(Bb + (long long)(tn * 64 +  0 + (lane & 15)) * sBn + bOff0);
    GCF* b1 = (GCF*)(Bb + (long long)(tn * 64 + 16 + (lane & 15)) * sBn + bOff0);
    GCF* b2 = (GCF*)(Bb + (long long)(tn * 64 + 32 + (lane & 15)) * sBn + bOff0);
    GCF* b3 = (GCF*)(Bb + (long long)(tn * 64 + 48 + (lane & 15)) * sBn + bOff0);
    const float* aPre = Ab + (long long)(tm * 32 + (lane & 15)) * sAm + kb;  // prefetch mirror
    long long bStep = CONTIG_B ? 32 : (long long)32 * sBk;

    v8f acc[2][4];
#pragma unroll
    for (int mi = 0; mi < 2; ++mi)
#pragma unroll
        for (int ni = 0; ni < 4; ++ni)
            acc[mi][ni] = (v8f){};

    for (int k0 = 0; k0 < K; k0 += 32) {
        // one light prefetch of the next A K-slab (global_prefetch_b8)
        if (k0 + 32 < K) __builtin_prefetch(aPre + 32, 0, 1);

        v16h af0 = load_frag_c(a0);
        v16h af1 = load_frag_c(a1);
        v16h bf0, bf1, bf2, bf3;
        if (CONTIG_B) {
            bf0 = load_frag_c(b0); bf1 = load_frag_c(b1);
            bf2 = load_frag_c(b2); bf3 = load_frag_c(b3);
        } else {
            bf0 = load_frag_s(b0, sBk); bf1 = load_frag_s(b1, sBk);
            bf2 = load_frag_s(b2, sBk); bf3 = load_frag_s(b3, sBk);
        }

        acc[0][0] = __builtin_amdgcn_wmma_f32_16x16x32_f16(false, af0, false, bf0, (short)0, acc[0][0], false, false);
        acc[0][1] = __builtin_amdgcn_wmma_f32_16x16x32_f16(false, af0, false, bf1, (short)0, acc[0][1], false, false);
        acc[0][2] = __builtin_amdgcn_wmma_f32_16x16x32_f16(false, af0, false, bf2, (short)0, acc[0][2], false, false);
        acc[0][3] = __builtin_amdgcn_wmma_f32_16x16x32_f16(false, af0, false, bf3, (short)0, acc[0][3], false, false);
        acc[1][0] = __builtin_amdgcn_wmma_f32_16x16x32_f16(false, af1, false, bf0, (short)0, acc[1][0], false, false);
        acc[1][1] = __builtin_amdgcn_wmma_f32_16x16x32_f16(false, af1, false, bf1, (short)0, acc[1][1], false, false);
        acc[1][2] = __builtin_amdgcn_wmma_f32_16x16x32_f16(false, af1, false, bf2, (short)0, acc[1][2], false, false);
        acc[1][3] = __builtin_amdgcn_wmma_f32_16x16x32_f16(false, af1, false, bf3, (short)0, acc[1][3], false, false);

        a0 += 32; a1 += 32; aPre += 32;
        b0 += bStep; b1 += bStep; b2 += bStep; b3 += bStep;
    }

    // Epilogue: C layout — VGPR r of a tile -> M = tileM + r + (lane>=16)*8, N = tileN + (lane&15)
    GF* Cg = (GF*)C;
#pragma unroll
    for (int ni = 0; ni < 4; ++ni) {
        int nn = tn * 64 + ni * 16 + (lane & 15);
        float bval = bias ? ((GCF*)bias)[nn] : 0.f;
#pragma unroll
        for (int mi = 0; mi < 2; ++mi) {
            int mb = tm * 32 + mi * 16 + ((lane >> 4) << 3);
#pragma unroll
            for (int r = 0; r < 8; ++r) {
                float v = acc[mi][ni][r] * alpha + bval;
                v = act_apply(v, act);
                Cg[offC + (long long)(mb + r) * sCm + nn] = v;
            }
        }
    }
}

// ---------------------------------------------------------------------------
// Naive GEMM for tiny N (sigma N=8, x_proj N=8, final N=38): thread per (m,n)
// ---------------------------------------------------------------------------
__global__ void small_gemm_kernel(const float* __restrict__ A, const float* __restrict__ W,
                                  const float* __restrict__ bias, float* __restrict__ C,
                                  int M, int N, int K, int act)
{
    int idx = blockIdx.x * blockDim.x + threadIdx.x;
    if (idx >= M * N) return;
    int m = idx / N, n = idx % N;
    float acc = bias ? bias[n] : 0.f;
    const float* a = A + (size_t)m * K;
    const float* w = W + (size_t)n * K;
    for (int k = 0; k < K; ++k) acc += a[k] * w[k];
    C[idx] = act_apply(acc, act);
}

// ---------------------------------------------------------------------------
// Embedding: circular conv1d (kernel 3) + sinusoidal positional encoding
// ---------------------------------------------------------------------------
__global__ void embed_kernel(const float* __restrict__ x, const float* __restrict__ tokw,
                             float* __restrict__ h, float* __restrict__ orig)
{
    int idx = blockIdx.x * blockDim.x + threadIdx.x;
    if (idx >= NB * LLEN * DMODEL) return;
    int d = idx % DMODEL;
    int l = (idx / DMODEL) % LLEN;
    int b = idx / (DMODEL * LLEN);
    int lm = (l == 0) ? LLEN - 1 : l - 1;
    int lp = (l == LLEN - 1) ? 0 : l + 1;
    float acc = 0.f;
    for (int c = 0; c < ENCIN; ++c) {
        const float* wv = tokw + ((size_t)d * ENCIN + c) * 3;
        acc += x[((size_t)b * LLEN + lm) * ENCIN + c] * wv[0]
             + x[((size_t)b * LLEN + l ) * ENCIN + c] * wv[1]
             + x[((size_t)b * LLEN + lp) * ENCIN + c] * wv[2];
    }
    int i2 = d & ~1;
    float div = expf((float)i2 * (-9.210340371976184f / (float)DMODEL)); // -ln(10000)/d
    float ang = (float)l * div;
    float pe = (d & 1) ? cosf(ang) : sinf(ang);
    float v = acc + pe;
    h[idx] = v;
    orig[idx] = v;
}

// ---------------------------------------------------------------------------
// In-place softmax over rows of length N (one block of 256 per row)
// ---------------------------------------------------------------------------
__global__ void softmax_kernel(float* __restrict__ buf, int N)
{
    __shared__ float red[256];
    int r = blockIdx.x;
    size_t base = (size_t)r * N;
    float mx = -1e30f;
    for (int j = threadIdx.x; j < N; j += 256) mx = fmaxf(mx, buf[base + j]);
    red[threadIdx.x] = mx; __syncthreads();
    for (int s = 128; s > 0; s >>= 1) {
        if (threadIdx.x < s) red[threadIdx.x] = fmaxf(red[threadIdx.x], red[threadIdx.x + s]);
        __syncthreads();
    }
    mx = red[0]; __syncthreads();
    float sum = 0.f;
    for (int j = threadIdx.x; j < N; j += 256) {
        float e = expf(buf[base + j] - mx);
        buf[base + j] = e; sum += e;
    }
    red[threadIdx.x] = sum; __syncthreads();
    for (int s = 128; s > 0; s >>= 1) {
        if (threadIdx.x < s) red[threadIdx.x] += red[threadIdx.x + s];
        __syncthreads();
    }
    float inv = 1.f / red[0];
    for (int j = threadIdx.x; j < N; j += 256) buf[base + j] *= inv;
}

// ---------------------------------------------------------------------------
// Gaussian prior + sigma broadcast. One block per (b,h,l) row.
// ---------------------------------------------------------------------------
__global__ void prior_sigma_kernel(const float* __restrict__ sig,
                                   float* __restrict__ prior, float* __restrict__ sigma)
{
    __shared__ float row[LLEN];
    __shared__ float red[256];
    int r = blockIdx.x;                 // r = (b*NH + h)*LLEN + l
    int l = r % LLEN;
    int bh = r / LLEN;
    int hh = bh % NH, bb = bh / NH;
    float sv = sig[((size_t)bb * LLEN + l) * NH + hh];
    float s = 1.f / (1.f + expf(-5.f * sv)) + 1e-5f;
    s = powf(3.f, s) - 1.f;
    float inv2s2 = 1.f / (2.f * s * s);
    float coef = 1.f / (2.5066282746310002f * s);   // sqrt(2*pi)*s
    float part = 0.f;
    for (int j = threadIdx.x; j < LLEN; j += 256) {
        float d = (float)(l - j);
        float p = expf(-d * d * inv2s2) * coef;
        row[j] = p; part += p;
    }
    red[threadIdx.x] = part; __syncthreads();
    for (int s2 = 128; s2 > 0; s2 >>= 1) {
        if (threadIdx.x < s2) red[threadIdx.x] += red[threadIdx.x + s2];
        __syncthreads();
    }
    float inv = 1.f / (red[0] + 1e-8f);
    size_t base = (size_t)r * LLEN;
    for (int j = threadIdx.x; j < LLEN; j += 256) {
        prior[base + j] = row[j] * inv;
        sigma[base + j] = s;
    }
}

// ---------------------------------------------------------------------------
// out = LayerNorm(a + b) * g + beta    (b may be null). One block per row.
// ---------------------------------------------------------------------------
__global__ void add_ln_kernel(const float* __restrict__ a, const float* __restrict__ b,
                              const float* __restrict__ g, const float* __restrict__ be,
                              float* __restrict__ out)
{
    __shared__ float r1[256], r2[256];
    int r = blockIdx.x;
    size_t base = (size_t)r * DMODEL;
    float s = 0.f, ss = 0.f;
    for (int j = threadIdx.x; j < DMODEL; j += 256) {
        float t = a[base + j] + (b ? b[base + j] : 0.f);
        s += t; ss += t * t;
    }
    r1[threadIdx.x] = s; r2[threadIdx.x] = ss; __syncthreads();
    for (int k = 128; k > 0; k >>= 1) {
        if (threadIdx.x < k) { r1[threadIdx.x] += r1[threadIdx.x + k]; r2[threadIdx.x] += r2[threadIdx.x + k]; }
        __syncthreads();
    }
    float mean = r1[0] / (float)DMODEL;
    float var  = r2[0] / (float)DMODEL - mean * mean;
    float inv  = rsqrtf(var + 1e-5f);
    for (int j = threadIdx.x; j < DMODEL; j += 256) {
        float t = a[base + j] + (b ? b[base + j] : 0.f);
        out[base + j] = (t - mean) * inv * g[j] + be[j];
    }
}

// ---------------------------------------------------------------------------
// h = orig = LayerNorm(g*xm + (1-g)*xo) * encg + encb
// ---------------------------------------------------------------------------
__global__ void gatemix_ln_kernel(const float* __restrict__ g, const float* __restrict__ xm,
                                  const float* __restrict__ xo,
                                  const float* __restrict__ gam, const float* __restrict__ bet,
                                  float* __restrict__ h, float* __restrict__ orig)
{
    __shared__ float r1[256], r2[256];
    int r = blockIdx.x;
    size_t base = (size_t)r * DMODEL;
    float s = 0.f, ss = 0.f;
    for (int j = threadIdx.x; j < DMODEL; j += 256) {
        float gv = g[base + j];
        float t = gv * xm[base + j] + (1.f - gv) * xo[base + j];
        s += t; ss += t * t;
    }
    r1[threadIdx.x] = s; r2[threadIdx.x] = ss; __syncthreads();
    for (int k = 128; k > 0; k >>= 1) {
        if (threadIdx.x < k) { r1[threadIdx.x] += r1[threadIdx.x + k]; r2[threadIdx.x] += r2[threadIdx.x + k]; }
        __syncthreads();
    }
    float mean = r1[0] / (float)DMODEL;
    float var  = r2[0] / (float)DMODEL - mean * mean;
    float inv  = rsqrtf(var + 1e-5f);
    for (int j = threadIdx.x; j < DMODEL; j += 256) {
        float gv = g[base + j];
        float t = gv * xm[base + j] + (1.f - gv) * xo[base + j];
        float o = (t - mean) * inv * gam[j] + bet[j];
        h[base + j] = o;
        orig[base + j] = o;
    }
}

// ---------------------------------------------------------------------------
// Mamba pieces
// ---------------------------------------------------------------------------
__global__ void silu_split_kernel(const float* __restrict__ xz, float* __restrict__ xs)
{
    int idx = blockIdx.x * blockDim.x + threadIdx.x;
    if (idx >= MROWS * DINNER) return;
    int m = idx / DINNER, c = idx % DINNER;
    float v = xz[(size_t)m * (2 * DINNER) + c];
    xs[idx] = v / (1.f + expf(-v));
}

__global__ void delta_kernel(const float* __restrict__ t8, const float* __restrict__ dtw,
                             const float* __restrict__ dtb, float* __restrict__ dl)
{
    int idx = blockIdx.x * blockDim.x + threadIdx.x;
    if (idx >= MROWS * DINNER) return;
    int m = idx / DINNER, c = idx % DINNER;
    float acc = dtb[c];
#pragma unroll
    for (int j = 0; j < DSTATE; ++j) acc += t8[(size_t)m * DSTATE + j] * dtw[(size_t)c * DSTATE + j];
    dl[idx] = (acc > 20.f) ? acc : log1pf(expf(acc));     // softplus
}

__global__ void mamba_scan_kernel(const float* __restrict__ xs, const float* __restrict__ dl,
                                  const float* __restrict__ A_log, const float* __restrict__ Dp,
                                  float* __restrict__ ys)
{
    int idx = blockIdx.x * blockDim.x + threadIdx.x;
    if (idx >= NB * DINNER) return;
    int c = idx % DINNER, b = idx / DINNER;
    float a[DSTATE], hs[DSTATE];
#pragma unroll
    for (int j = 0; j < DSTATE; ++j) { a[j] = -expf(A_log[(size_t)c * DSTATE + j]); hs[j] = 0.f; }
    float dp = Dp[c];
    for (int t = 0; t < LLEN; ++t) {
        size_t o = ((size_t)b * LLEN + t) * DINNER + c;
        float xt = xs[o], dt = dl[o];
        float y = 0.f;
#pragma unroll
        for (int j = 0; j < DSTATE; ++j) {
            hs[j] = expf(dt * a[j]) * hs[j] + dt * xt;
            y += hs[j];
        }
        ys[o] = y + dp * xt;
    }
}

__global__ void ymul_kernel(const float* __restrict__ xz, float* __restrict__ ys)
{
    int idx = blockIdx.x * blockDim.x + threadIdx.x;
    if (idx >= MROWS * DINNER) return;
    int m = idx / DINNER, c = idx % DINNER;
    float z = xz[(size_t)m * (2 * DINNER) + DINNER + c];
    ys[idx] *= z / (1.f + expf(-z));
}

__global__ void add_kernel(float* __restrict__ dst, const float* __restrict__ src, int n)
{
    int i = blockIdx.x * blockDim.x + threadIdx.x;
    if (i < n) dst[i] += src[i];
}

__global__ void cat_kernel(const float* __restrict__ a, const float* __restrict__ b,
                           float* __restrict__ c)
{
    int idx = blockIdx.x * blockDim.x + threadIdx.x;
    if (idx >= MROWS * 2 * DMODEL) return;
    int m = idx / (2 * DMODEL), j = idx % (2 * DMODEL);
    c[idx] = (j < DMODEL) ? a[(size_t)m * DMODEL + j] : b[(size_t)m * DMODEL + (j - DMODEL)];
}

// ---------------------------------------------------------------------------
// Host-side helpers
// ---------------------------------------------------------------------------
static inline void gemm(hipStream_t st, const float* A, const float* B, const float* bias,
                        float* C, int M, int N, int K,
                        int sAm, int sBn, int sBk, int sCm,
                        long long sAo, long long sAi, long long sBo, long long sBi,
                        long long sCo, long long sCi, int zdiv, int nbatch,
                        float alpha, int act)
{
    dim3 grid(N / 64, M / 32, nbatch);
    if (sBk == 1)
        wmma_gemm_kernel<true><<<grid, 32, 0, st>>>(A, B, bias, C, K, sAm, sBn, sBk, sCm,
                                                    sAo, sAi, sBo, sBi, sCo, sCi, zdiv, alpha, act);
    else
        wmma_gemm_kernel<false><<<grid, 32, 0, st>>>(A, B, bias, C, K, sAm, sBn, sBk, sCm,
                                                     sAo, sAi, sBo, sBi, sCo, sCi, zdiv, alpha, act);
}

static inline void gemm_plain(hipStream_t st, const float* A, const float* W, const float* bias,
                              float* C, int M, int N, int K, int act)
{
    gemm(st, A, W, bias, C, M, N, K, K, K, 1, N, 0, 0, 0, 0, 0, 0, 1, 1, 1.f, act);
}

extern "C" void kernel_launch(void* const* d_in, const int* in_sizes, int n_in,
                              void* d_out_, int out_size, void* d_ws, size_t ws_size,
                              hipStream_t stream)
{
    const float* x     = (const float*)d_in[0];
    const float* tokw  = (const float*)d_in[1];
    const float* Wq    = (const float*)d_in[2];
    const float* bq    = (const float*)d_in[3];
    const float* Wk    = (const float*)d_in[4];
    const float* bk    = (const float*)d_in[5];
    const float* Wv    = (const float*)d_in[6];
    const float* bvv   = (const float*)d_in[7];
    const float* Wsig  = (const float*)d_in[8];
    const float* bsig  = (const float*)d_in[9];
    const float* Wo    = (const float*)d_in[10];
    const float* bo    = (const float*)d_in[11];
    const float* c1w   = (const float*)d_in[12];
    const float* c1b   = (const float*)d_in[13];
    const float* c2w   = (const float*)d_in[14];
    const float* c2b   = (const float*)d_in[15];
    const float* ln1g  = (const float*)d_in[16];
    const float* ln1b  = (const float*)d_in[17];
    const float* ln2g  = (const float*)d_in[18];
    const float* ln2b  = (const float*)d_in[19];
    const float* inpw  = (const float*)d_in[20];
    const float* A_log = (const float*)d_in[21];
    const float* Dp    = (const float*)d_in[22];
    const float* xpw   = (const float*)d_in[23];
    const float* dtw   = (const float*)d_in[24];
    const float* dtb   = (const float*)d_in[25];
    const float* moutw = (const float*)d_in[26];
    const float* gatew = (const float*)d_in[27];
    const float* gateb = (const float*)d_in[28];
    const float* encg  = (const float*)d_in[29];
    const float* encb  = (const float*)d_in[30];
    const float* projw = (const float*)d_in[31];
    const float* projb = (const float*)d_in[32];

    float* outp = (float*)d_out_;
    float* ws   = (float*)d_ws;

    const size_t SZ = (size_t)MROWS * DMODEL;
    float* S_H  = ws;               float* S_OR = S_H  + SZ;
    float* S_Q  = S_OR + SZ;        float* S_K  = S_Q  + SZ;
    float* S_V  = S_K  + SZ;        float* S_A  = S_V  + SZ;
    float* S_T  = S_A  + SZ;        float* S_X1 = S_T  + SZ;
    float* S_XO = S_X1 + SZ;        float* S_XM = S_XO + SZ;
    float* S_G  = S_XM + SZ;
    float* S_XZ = S_G  + SZ;                                  // MROWS * 1536
    float* S_XS = S_XZ + (size_t)MROWS * 2 * DINNER;          // MROWS * 768
    float* S_DL = S_XS + (size_t)MROWS * DINNER;
    float* S_YS = S_DL + (size_t)MROWS * DINNER;
    float* S_CT = S_YS + (size_t)MROWS * DINNER;              // MROWS * 1024
    float* S_SG = S_CT + (size_t)MROWS * 2 * DMODEL;          // MROWS * 8
    float* S_T8 = S_SG + (size_t)MROWS * NH;

    const size_t BHLL  = (size_t)NB * NH * LLEN * LLEN;       // 16,777,216
    const size_t O_SER = (size_t)NB * LLEN * COUT;            // out comes first
    const size_t O_PRI = O_SER + (size_t)NLAY * BHLL;
    const size_t O_SIG = O_PRI + (size_t)NLAY * BHLL;

    // ---- embedding + positional encoding -> h, original
    {
        int n = NB * LLEN * DMODEL;
        embed_kernel<<<(n + 255) / 256, 256, 0, stream>>>(x, tokw, S_H, S_OR);
    }

    for (int i = 0; i < NLAY; ++i) {
        const float* Wq_i  = Wq  + (size_t)i * DMODEL * DMODEL;
        const float* bq_i  = bq  + (size_t)i * DMODEL;
        const float* Wk_i  = Wk  + (size_t)i * DMODEL * DMODEL;
        const float* bk_i  = bk  + (size_t)i * DMODEL;
        const float* Wv_i  = Wv  + (size_t)i * DMODEL * DMODEL;
        const float* bv_i  = bvv + (size_t)i * DMODEL;
        const float* Ws_i  = Wsig + (size_t)i * NH * DMODEL;
        const float* bs_i  = bsig + (size_t)i * NH;
        const float* Wo_i  = Wo  + (size_t)i * DMODEL * DMODEL;
        const float* bo_i  = bo  + (size_t)i * DMODEL;
        const float* c1w_i = c1w + (size_t)i * DMODEL * DMODEL;
        const float* c1b_i = c1b + (size_t)i * DMODEL;
        const float* c2w_i = c2w + (size_t)i * DMODEL * DMODEL;
        const float* c2b_i = c2b + (size_t)i * DMODEL;
        const float* l1g_i = ln1g + (size_t)i * DMODEL;
        const float* l1b_i = ln1b + (size_t)i * DMODEL;
        const float* l2g_i = ln2g + (size_t)i * DMODEL;
        const float* l2b_i = ln2b + (size_t)i * DMODEL;

        // Q, K, V projections (WMMA)
        gemm_plain(stream, S_H, Wq_i, bq_i, S_Q, MROWS, DMODEL, DMODEL, 0);
        gemm_plain(stream, S_H, Wk_i, bk_i, S_K, MROWS, DMODEL, DMODEL, 0);
        gemm_plain(stream, S_H, Wv_i, bv_i, S_V, MROWS, DMODEL, DMODEL, 0);

        // sigma projection (N=8, naive)
        {
            int n = MROWS * NH;
            small_gemm_kernel<<<(n + 255) / 256, 256, 0, stream>>>(S_H, Ws_i, bs_i, S_SG,
                                                                   MROWS, NH, DMODEL, 0);
        }

        // scores -> series region of d_out (batched per (b,h), alpha = 1/sqrt(64))
        float* ser = outp + O_SER + (size_t)i * BHLL;
        gemm(stream, S_Q, S_K, nullptr, ser,
             LLEN, LLEN, EHD,
             DMODEL, DMODEL, 1, LLEN,
             (long long)LLEN * DMODEL, EHD,
             (long long)LLEN * DMODEL, EHD,
             (long long)NH * LLEN * LLEN, (long long)LLEN * LLEN,
             NH, NB * NH, 0.125f, 0);

        softmax_kernel<<<NB * NH * LLEN, 256, 0, stream>>>(ser, LLEN);

        // prior + sigma outputs
        prior_sigma_kernel<<<NB * NH * LLEN, 256, 0, stream>>>(
            S_SG, outp + O_PRI + (size_t)i * BHLL, outp + O_SIG + (size_t)i * BHLL);

        // attn_out = series @ V   (batched per (b,h); B is K-major: sBn=1, sBk=512)
        gemm(stream, ser, S_V, nullptr, S_A,
             LLEN, EHD, LLEN,
             LLEN, 1, DMODEL, DMODEL,
             (long long)NH * LLEN * LLEN, (long long)LLEN * LLEN,
             (long long)LLEN * DMODEL, EHD,
             (long long)LLEN * DMODEL, EHD,
             NH, NB * NH, 1.f, 0);

        // output projection + residual + LN1
        gemm_plain(stream, S_A, Wo_i, bo_i, S_T, MROWS, DMODEL, DMODEL, 0);
        add_ln_kernel<<<MROWS, 256, 0, stream>>>(S_H, S_T, l1g_i, l1b_i, S_X1);

        // FFN: gelu GEMM then linear GEMM, residual + LN2
        gemm_plain(stream, S_X1, c1w_i, c1b_i, S_Q, MROWS, DMODEL, DMODEL, 1);  // gelu, D_FF=512
        gemm_plain(stream, S_Q, c2w_i, c2b_i, S_T, MROWS, DMODEL, DMODEL, 0);
        add_ln_kernel<<<MROWS, 256, 0, stream>>>(S_X1, S_T, l2g_i, l2b_i, S_XO);

        // ---- Mamba block ----
        gemm_plain(stream, S_XO, inpw, nullptr, S_XZ, MROWS, 2 * DINNER, DMODEL, 0);
        {
            int n = MROWS * DINNER;
            silu_split_kernel<<<(n + 255) / 256, 256, 0, stream>>>(S_XZ, S_XS);
        }
        {
            int n = MROWS * DSTATE;
            small_gemm_kernel<<<(n + 255) / 256, 256, 0, stream>>>(S_XS, xpw, nullptr, S_T8,
                                                                   MROWS, DSTATE, DINNER, 0);
        }
        {
            int n = MROWS * DINNER;
            delta_kernel<<<(n + 255) / 256, 256, 0, stream>>>(S_T8, dtw, dtb, S_DL);
        }
        {
            int n = NB * DINNER;
            mamba_scan_kernel<<<(n + 255) / 256, 256, 0, stream>>>(S_XS, S_DL, A_log, Dp, S_YS);
        }
        {
            int n = MROWS * DINNER;
            ymul_kernel<<<(n + 255) / 256, 256, 0, stream>>>(S_XZ, S_YS);
        }
        gemm_plain(stream, S_YS, moutw, nullptr, S_XM, MROWS, DMODEL, DINNER, 0);
        {
            int n = MROWS * DMODEL;
            add_kernel<<<(n + 255) / 256, 256, 0, stream>>>(S_XM, S_OR, n);  // + original
        }

        // gate = sigmoid([xo, xm] @ gate_w.T + gate_b)
        {
            int n = MROWS * 2 * DMODEL;
            cat_kernel<<<(n + 255) / 256, 256, 0, stream>>>(S_XO, S_XM, S_CT);
        }
        gemm_plain(stream, S_CT, gatew, gateb, S_G, MROWS, DMODEL, 2 * DMODEL, 3);  // sigmoid

        // h = original = LN(g*xm + (1-g)*xo)
        gatemix_ln_kernel<<<MROWS, 256, 0, stream>>>(S_G, S_XM, S_XO, encg, encb, S_H, S_OR);
    }

    // final projection (N=38, naive) -> first out_size region of d_out
    {
        int n = MROWS * COUT;
        small_gemm_kernel<<<(n + 255) / 256, 256, 0, stream>>>(S_H, projw, projb, outp,
                                                               MROWS, COUT, DMODEL, 0);
    }
}
